// SpectralConv2d_88862873354312
// MI455X (gfx1250) — compile-verified, matrix operand order checked by reading, not run on hardware
//
#include <hip/hip_runtime.h>

// SpectralConv2d (FNO layer) for gfx1250 / MI455X.
//
// The FFT truncation to 32x32 modes turns the whole op into four dense-GEMM
// stages against fixed DFT bases; all GEMMs run on v_wmma_f32_16x16x32_bf16
// (f32 accumulate).  Memory-bound: stream x once (268 MB), out once (268 MB);
// 2x8 MB of spectral blocks stay L2-resident (192 MB L2).
//
//   K0: bf16 DFT basis tables, PRE-SWIZZLED into WMMA fragment layout.
//   K1: per (b,ci) image: P = X @ [cos|-sin]  (256x256 @ 256x64),
//       block = A2 @ stacked(P)               (64x512 @ 512x32) -> ws1
//       (x-stream staged via async global->LDS, double-buffered, if available)
//   K2: per mode (1024): complex GEMM [16x128]@[128x64] over channels -> ws2
//   K3: per (b,co) image: Y = A3 @ B3 (256x64 @ 64x64),
//       out = Y @ W4 (256x64 @ 64x256) -> d_out

typedef __bf16 bf;
typedef __attribute__((ext_vector_type(16))) __bf16 v16bf;
typedef __attribute__((ext_vector_type(8)))  __bf16 v8bf;
typedef __attribute__((ext_vector_type(8)))  float  v8f;
typedef __attribute__((ext_vector_type(4)))  int    v4i;

#define WMMA_BF16(a, b, c) \
  __builtin_amdgcn_wmma_f32_16x16x32_bf16(false, (a), false, (b), (short)0, (c), false, false)

#if __has_builtin(__builtin_amdgcn_global_load_async_to_lds_b128) && \
    __has_builtin(__builtin_amdgcn_s_wait_asynccnt)
#define HAVE_ASYNC_LDS 1
typedef __attribute__((address_space(1))) v4i gv4i;   // global v4i
typedef __attribute__((address_space(3))) v4i lv4i;   // LDS v4i
#endif

// Native f32->bf16 (gfx1250 has bf16 cvt hardware; avoids 4-op RNE emulation)
__device__ __forceinline__ bf f2bf(float f) { return (bf)f; }

__device__ __forceinline__ bf bneg(bf v) {
  unsigned short u = __builtin_bit_cast(unsigned short, v);
  u ^= 0x8000u;
  return __builtin_bit_cast(bf, u);
}

// ---- fragment helpers (wave32 layouts per CDNA5 ISA 7.12.2) -------------
// A (16x32 bf16): lane L: m=L&15, hi=L>>4; elem e -> k = e + (e&8) + hi*8
// B (32x16 bf16): lane L: n=L&15, hi=L>>4; same k pattern
// C/D (16x16 f32): lane L: n=L&15; vgpr v -> m = v + 8*(L>>4)

__device__ __forceinline__ v16bf frag_tab(const bf* tileBase) {
  return ((const v16bf*)tileBase)[threadIdx.x & 31];
}

__device__ __forceinline__ v16bf a_frag_lds(const bf* lds, int r0, int c0, int stride) {
  int lane = threadIdx.x & 31, m = lane & 15, hi = lane >> 4;
  const v8bf* p0 = (const v8bf*)&lds[(r0 + m) * stride + c0 + hi * 8];
  const v8bf* p1 = (const v8bf*)&lds[(r0 + m) * stride + c0 + hi * 8 + 16];
  v8bf x0 = *p0, x1 = *p1;
  v16bf r;
#pragma unroll
  for (int e = 0; e < 8; ++e) { r[e] = x0[e]; r[8 + e] = x1[e]; }
  return r;
}

// A fragment straight from global f32 (fallback x-stream path)
__device__ __forceinline__ v16bf a_frag_gx(const float* X, int r0, int c0) {
  int lane = threadIdx.x & 31, m = lane & 15, hi = lane >> 4;
  const float* p = X + (size_t)(r0 + m) * 256 + c0 + hi * 8;
  __builtin_prefetch(p + 32, 0, 1);
  v16bf r;
#pragma unroll
  for (int e = 0; e < 8; ++e) r[e] = f2bf(p[e]);
#pragma unroll
  for (int e = 0; e < 8; ++e) r[8 + e] = f2bf(p[16 + e]);
  return r;
}

// A fragment from an LDS-resident f32 panel [16 rows x 32 cols]
__device__ __forceinline__ v16bf a_frag_panel(const float* pan) {
  int lane = threadIdx.x & 31, m = lane & 15, hi = lane >> 4;
  const float* q = pan + m * 32 + hi * 8;
  v16bf r;
#pragma unroll
  for (int e = 0; e < 8; ++e) r[e] = f2bf(q[e]);
#pragma unroll
  for (int e = 0; e < 8; ++e) r[8 + e] = f2bf(q[16 + e]);
  return r;
}

__device__ __forceinline__ v16bf b_frag_lds(const bf* lds, int k0, int n0, int stride) {
  int lane = threadIdx.x & 31, n = lane & 15, hi = lane >> 4;
  v16bf r;
#pragma unroll
  for (int e = 0; e < 16; ++e) {
    int k = e + (e & 8) + hi * 8;
    r[e] = lds[(k0 + k) * stride + n0 + n];
  }
  return r;
}

__device__ __forceinline__ void store_frag_f32(float* dst, int r0, int c0, int stride, v8f c) {
  int lane = threadIdx.x & 31, n = lane & 15, hi = lane >> 4;
#pragma unroll
  for (int v = 0; v < 8; ++v)
    dst[(size_t)(r0 + v + 8 * hi) * stride + c0 + n] = c[v];
}
__device__ __forceinline__ void store_frag_bf16(bf* dst, int r0, int c0, int stride, v8f c) {
  int lane = threadIdx.x & 31, n = lane & 15, hi = lane >> 4;
#pragma unroll
  for (int v = 0; v < 8; ++v)
    dst[(r0 + v + 8 * hi) * stride + c0 + n] = f2bf(c[v]);
}

#ifdef HAVE_ASYNC_LDS
// Stage a 16x32 f32 panel (2 KB) of X into LDS: 4 async b128 ops per wave
// (each op: 32 lanes x 16 B).  Tracked with ASYNCcnt.
__device__ __forceinline__ void stage_panel_async(const float* X, int r0, int c0,
                                                  float* ldsDst) {
  int lane = threadIdx.x & 31;
#pragma unroll
  for (int it = 0; it < 4; ++it) {
    int chunk = it * 32 + lane;          // 128 chunks of 16 B
    int row = chunk >> 3, c16 = chunk & 7;
    const float* g = X + (size_t)(r0 + row) * 256 + c0 + c16 * 4;
    float* l = ldsDst + row * 32 + c16 * 4;
    gv4i* gp = (gv4i*)(unsigned long long)(__SIZE_TYPE__)g;
    lv4i* lp = (lv4i*)(unsigned)(__SIZE_TYPE__)l;
    __builtin_amdgcn_global_load_async_to_lds_b128(gp, lp, 0, 0);
  }
}
#endif

// ---- K0: DFT basis tables, pre-swizzled -----------------------------------
// tile map (each tile = 512 bf16, lane-major: lane*16 + e):
//   T1 [  0, 32): stage1 B  [256x64]  tile = kt*4  + nt   (B layout)
//   T2 [ 32, 96): stage2 A  [ 64x512] tile = mt*16 + kt   (A layout)
//   T3 [ 96,128): stage3A A [256x64]  tile = mt*2  + kt   (A layout)
//   T4 [128,160): stage3B B [ 64x256] tile = kt*16 + nt   (B layout)
__global__ __launch_bounds__(256) void k0_tables(bf* __restrict__ tab) {
  const float W0 = 6.283185307179586f / 256.0f;
  int tile = blockIdx.x;
#pragma unroll
  for (int half = 0; half < 2; ++half) {
    int idx  = threadIdx.x + half * 256;     // 0..511 within tile
    int lane = idx >> 4, e = idx & 15;
    int hi = lane >> 4, lo = lane & 15;
    int klocal = e + (e & 8) + hi * 8;       // K within 32-wide tile
    float val = 0.0f;
    if (tile < 32) {                          // T1: W1[K=w][N]
      int kt = tile >> 2, nt = tile & 3;
      int K = kt * 32 + klocal, N = nt * 16 + lo;
      int j = (N < 32) ? N : N - 32;
      float ang = (float)((K * j) & 255) * W0;
      val = (N < 32) ? __cosf(ang) : -__sinf(ang);
    } else if (tile < 96) {                   // T2: A2[M][K]
      int t = tile - 32; int mt = t >> 4, kt = t & 15;
      int M = mt * 16 + lo, K = kt * 32 + klocal;
      int k1 = (M < 32) ? M : M - 32;
      int h  = (K < 256) ? K : K - 256;
      float ang = (float)((h * k1) & 255) * W0;
      float cv = __cosf(ang), sv = __sinf(ang);
      val = (M < 32) ? ((K < 256) ? cv : sv)
                     : ((K < 256) ? -sv : cv);
    } else if (tile < 128) {                  // T3: A3[M=h][K]
      int t = tile - 96; int mt = t >> 1, kt = t & 1;
      int M = mt * 16 + lo, K = kt * 32 + klocal;
      int k1 = (K < 32) ? K : K - 32;
      float ang = (float)((M * k1) & 255) * W0;
      val = ((K < 32) ? __cosf(ang) : __sinf(ang)) * (1.0f / 256.0f);
    } else {                                  // T4: W4[K][N=w]
      int t = tile - 128; int kt = t >> 4, nt = t & 15;
      int K = kt * 32 + klocal, N = nt * 16 + lo;
      if (K < 32) {
        float ang = (float)((N * K) & 255) * W0;
        val = ((K == 0) ? 1.0f : 2.0f * __cosf(ang)) * (1.0f / 256.0f);
      } else {
        int j = K - 32;
        float ang = (float)((N * j) & 255) * W0;
        val = ((j == 0) ? 0.0f : -2.0f * __sinf(ang)) * (1.0f / 256.0f);
      }
    }
    tab[(size_t)tile * 512 + idx] = f2bf(val);
  }
}

// ---- K1: forward transform, one WG per (b,ci) image ----------------------
// Stage1: P[256x64] = X @ W1   (Re cols 0..31, Im cols 32..63)
// Stage2: block[64x32] = A2 @ stack(P)  (rows 0..31 Re, 32..63 Im) -> ws1
__global__ __launch_bounds__(256) void k1_forward(const float* __restrict__ x,
                                                  const bf* __restrict__ tab,
                                                  float* __restrict__ blocks) {
  __shared__ __align__(16) bf P[256 * 64];          // 32 KB
#ifdef HAVE_ASYNC_LDS
  __shared__ __align__(16) float Xpan[8][2][16 * 32]; // 32 KB, per-wave dbl-buf
#endif
  int img  = blockIdx.x;                     // b*64 + ci
  int wave = threadIdx.x >> 5;
  const float* X = x + (size_t)img * 65536;

  // stage 1: each wave owns 2 M-tiles (32 rows), all 4 N-tiles
#pragma unroll
  for (int mi = 0; mi < 2; ++mi) {
    int mt = wave * 2 + mi;
    v8f acc[4] = {v8f{}, v8f{}, v8f{}, v8f{}};
#ifdef HAVE_ASYNC_LDS
    stage_panel_async(X, mt * 16, 0, Xpan[wave][0]);
#pragma unroll
    for (int kt = 0; kt < 8; ++kt) {
      if (kt < 7) {
        stage_panel_async(X, mt * 16, (kt + 1) * 32, Xpan[wave][(kt + 1) & 1]);
        __builtin_amdgcn_s_wait_asynccnt(4);  // panel kt landed, kt+1 in flight
      } else {
        __builtin_amdgcn_s_wait_asynccnt(0);
      }
      v16bf aF = a_frag_panel(Xpan[wave][kt & 1]);
#pragma unroll
      for (int nt = 0; nt < 4; ++nt) {
        v16bf bF = frag_tab(tab + (size_t)(kt * 4 + nt) * 512);
        acc[nt] = WMMA_BF16(aF, bF, acc[nt]);
      }
    }
#else
    for (int kt = 0; kt < 8; ++kt) {
      v16bf aF = a_frag_gx(X, mt * 16, kt * 32);
#pragma unroll
      for (int nt = 0; nt < 4; ++nt) {
        v16bf bF = frag_tab(tab + (size_t)(kt * 4 + nt) * 512);
        acc[nt] = WMMA_BF16(aF, bF, acc[nt]);
      }
    }
#endif
#pragma unroll
    for (int nt = 0; nt < 4; ++nt)
      store_frag_bf16(P, mt * 16, nt * 16, 64, acc[nt]);
  }
  __syncthreads();

  // stage 2: output [64x32]; 8 tiles -> one per wave
  int mt2 = wave >> 1, nt2 = wave & 1;
  v8f acc = {};
  for (int kt = 0; kt < 16; ++kt) {
    v16bf aF = frag_tab(tab + (size_t)(32 + mt2 * 16 + kt) * 512);
    int k0 = (kt & 7) * 32;                         // h tile
    int n0 = nt2 * 16 + ((kt >= 8) ? 32 : 0);       // Re / Im halves of P
    v16bf bF = b_frag_lds(P, k0, n0, 64);
    acc = WMMA_BF16(aF, bF, acc);
  }
  store_frag_f32(blocks + (size_t)img * 2048, mt2 * 16, nt2 * 16, 32, acc);
}

// ---- K2: per-mode complex channel mixing, one WG per (k1,k2) -------------
// A[16x128] = [Xr | Xi] over (b, ci);  out_re: B=[Wr; -Wi], out_im: B=[Wi; Wr]
__global__ __launch_bounds__(256) void k2_mix(const float* __restrict__ wr,
                                              const float* __restrict__ wi,
                                              const float* __restrict__ bin,
                                              float* __restrict__ bout) {
  __shared__ __align__(16) bf A[16 * 128];   // 4 KB
  __shared__ __align__(16) bf Wrs[64 * 64];  // 8 KB
  __shared__ __align__(16) bf Wis[64 * 64];  // 8 KB
  int mode = blockIdx.x;                     // k1*32 + k2

  for (int t = threadIdx.x; t < 2048; t += 256) {
    int b = t >> 7, j = t & 127, i = j & 63, reim = j >> 6;
    A[b * 128 + j] = f2bf(bin[(size_t)(b * 64 + i) * 2048 + reim * 1024 + mode]);
  }
  for (int t = threadIdx.x; t < 4096; t += 256) {
    int i = t >> 6, o = t & 63;
    size_t widx = (size_t)(i * 64 + o) * 1024 + mode;
    Wrs[t] = f2bf(wr[widx]);
    Wis[t] = f2bf(wi[widx]);
  }
  __syncthreads();

  int wave = threadIdx.x >> 5;
  int re_out = wave >> 2, nt = wave & 3;
  int lane = threadIdx.x & 31, n = lane & 15, hi = lane >> 4;
  v8f acc = {};
  for (int kt = 0; kt < 4; ++kt) {
    v16bf aF = a_frag_lds(A, 0, kt * 32, 128);
    v16bf bF;
#pragma unroll
    for (int e = 0; e < 16; ++e) {
      int k = kt * 32 + e + (e & 8) + hi * 8;
      int c = nt * 16 + n;
      bf v;
      if (re_out == 0) v = (k < 64) ? Wrs[k * 64 + c] : bneg(Wis[(k - 64) * 64 + c]);
      else             v = (k < 64) ? Wis[k * 64 + c] : Wrs[(k - 64) * 64 + c];
      bF[e] = v;
    }
    acc = WMMA_BF16(aF, bF, acc);
  }
#pragma unroll
  for (int v = 0; v < 8; ++v) {
    int b = v + 8 * hi;             // D row = batch
    int o = nt * 16 + n;            // D col = cout
    bout[(size_t)(b * 64 + o) * 2048 + re_out * 1024 + mode] = acc[v];
  }
}

// ---- K3: inverse transform, one WG per (b,co) image ----------------------
// Stage A: Y[256x64] = A3 @ B3,  B3 = [[br, bi], [-bi, br]]
// Stage B: out[256x256] = Y @ W4
__global__ __launch_bounds__(256) void k3_inverse(const float* __restrict__ blocks,
                                                  const bf* __restrict__ tab,
                                                  float* __restrict__ out) {
  __shared__ __align__(16) bf B3[64 * 64];    // 8 KB
  __shared__ __align__(16) bf Y[256 * 64];    // 32 KB
  int img  = blockIdx.x;                      // b*64 + co
  int wave = threadIdx.x >> 5;
  const float* blk = blocks + (size_t)img * 2048;

  for (int t = threadIdx.x; t < 4096; t += 256) {
    int r = t >> 6, c = t & 63;
    float v;
    if (r < 32) v = (c < 32) ?  blk[r * 32 + c]               : blk[1024 + r * 32 + (c - 32)];
    else        v = (c < 32) ? -blk[1024 + (r - 32) * 32 + c] : blk[(r - 32) * 32 + (c - 32)];
    B3[t] = f2bf(v);
  }
  __syncthreads();

  // Stage A: [256x64] output; wave owns 2 M-tiles x 4 N-tiles
#pragma unroll
  for (int mi = 0; mi < 2; ++mi) {
    int mt = wave * 2 + mi;
#pragma unroll
    for (int nt = 0; nt < 4; ++nt) {
      v8f acc = {};
#pragma unroll
      for (int kt = 0; kt < 2; ++kt) {
        v16bf aF = frag_tab(tab + (size_t)(96 + mt * 2 + kt) * 512);
        v16bf bF = b_frag_lds(B3, kt * 32, nt * 16, 64);
        acc = WMMA_BF16(aF, bF, acc);
      }
      store_frag_bf16(Y, mt * 16, nt * 16, 64, acc);
    }
  }
  __syncthreads();

  // Stage B: [256x256] output; wave owns 2 M-tiles x 16 N-tiles
  float* O = out + (size_t)img * 65536;
#pragma unroll
  for (int mi = 0; mi < 2; ++mi) {
    int mt = wave * 2 + mi;
    v16bf aF0 = a_frag_lds(Y, mt * 16, 0, 64);
    v16bf aF1 = a_frag_lds(Y, mt * 16, 32, 64);
    for (int nt = 0; nt < 16; ++nt) {
      v8f acc = {};
      acc = WMMA_BF16(aF0, frag_tab(tab + (size_t)(128 + nt) * 512), acc);
      acc = WMMA_BF16(aF1, frag_tab(tab + (size_t)(128 + 16 + nt) * 512), acc);
      store_frag_f32(O, mt * 16, nt * 16, 256, acc);
    }
  }
}

// ---- host launch ----------------------------------------------------------
extern "C" void kernel_launch(void* const* d_in, const int* in_sizes, int n_in,
                              void* d_out, int out_size, void* d_ws, size_t ws_size,
                              hipStream_t stream) {
  const float* x  = (const float*)d_in[0];
  const float* wr = (const float*)d_in[1];
  const float* wi = (const float*)d_in[2];
  float* out = (float*)d_out;

  char*  ws  = (char*)d_ws;
  bf*    tab = (bf*)ws;                                // 160 KB tables
  float* ws1 = (float*)(ws + (256 << 10));             // 8 MB forward blocks
  float* ws2 = (float*)(ws + (256 << 10) + (8 << 20)); // 8 MB mixed blocks

  k0_tables <<<160,  256, 0, stream>>>(tab);
  k1_forward<<<1024, 256, 0, stream>>>(x, tab, ws1);
  k2_mix    <<<1024, 256, 0, stream>>>(wr, wi, ws1, ws2);
  k3_inverse<<<1024, 256, 0, stream>>>(ws2, tab, out);
}
